// RI_encoder_81475529605151
// MI455X (gfx1250) — matrix operand order repaired and checked by reading, not compiled, not run on hardware
//
#include <hip/hip_runtime.h>
#include <hip/hip_bf16.h>
#include <math.h>

typedef __attribute__((ext_vector_type(16))) __bf16 v16bf;
typedef __attribute__((ext_vector_type(8)))  __bf16 v8bf;
typedef __attribute__((ext_vector_type(8)))  float  v8f;

#define PIXELS 262144ull   // B(16) * ns(32) * S(512)
#define NSAMP  8192        // N points per batch in pts / norm_ori
#define EPSF   1e-7f
#define EPSBN  1e-5f

// ---------------------------------------------------------------------------
// Stage 1: rotation-invariant features. One wave32 per (b,s) neighborhood.
// Writes X0 as bf16, PIXEL-major [P][32] (channels 8..31 zero-padded so the
// first GEMM runs K=32 bf16 WMMA). Each lane stores 64 contiguous bytes.
// ---------------------------------------------------------------------------
__global__ __launch_bounds__(128)
void ri_kernel(const float* __restrict__ xyz, const float* __restrict__ centers,
               const float* __restrict__ cnorms, const int* __restrict__ idx,
               const float* __restrict__ pts, const float* __restrict__ norm_ori,
               __bf16* __restrict__ X0)
{
    const int lane = threadIdx.x & 31;
    const int g = blockIdx.x * 4 + (threadIdx.x >> 5);   // group id in [0, 8192)
    const int b = g >> 9;
    const int s = g & 511;

    const float* vv = xyz + ((size_t)g * 32 + lane) * 3;
    float vx = vv[0], vy = vv[1], vz = vv[2];
    float nx = cnorms[g*3+0], ny = cnorms[g*3+1], nz = cnorms[g*3+2];
    float cx = centers[g*3+0], cy = centers[g*3+1], cz = centers[g*3+2];
    int   iv = idx[(size_t)g * 32 + lane];

    // projection onto plane normal to nnorm
    float dp = vx*nx + vy*ny + vz*nz;
    float px = vx - dp*nx, py = vy - dp*ny, pz = vz - dp*nz;
    float plen = sqrtf(px*px + py*py + pz*pz);
    float pinv = (plen == 0.f) ? 1.f : plen;
    float ux = px/pinv, uy = py/pinv, uz = pz/pinv;

    // wave argmax of projected length (first max wins, like jnp.argmax)
    float mbest = -INFINITY; int mi = 0;
    for (int j = 0; j < 32; ++j) {
        float pj = __shfl(plen, j, 32);
        if (pj > mbest) { mbest = pj; mi = j; }
    }
    float rx = __shfl(ux, mi, 32), ry = __shfl(uy, mi, 32), rz = __shfl(uz, mi, 32);

    float d = ux*rx + uy*ry + uz*rz;
    float crx = uy*rz - uz*ry, cry = uz*rx - ux*rz, crz = ux*ry - uy*rx;
    float cd  = crx*nx + cry*ny + crz*nz;
    float sg  = (cd > 0.f) ? 1.f : ((cd < 0.f) ? -1.f : 0.f);
    if (lane == 0) sg = 1.f;
    d = sg * d - (1.f - sg);

    // stable descending rank of d across the wave
    int rnk = 0;
    for (int j = 0; j < 32; ++j) {
        float dj = __shfl(d, j, 32);
        rnk += (dj > d) || (dj == d && j < lane);
    }
    // inverse permutation: which source lane lands at sorted position 'lane'
    int src = 0;
    for (int j = 0; j < 32; ++j) {
        int rj = __shfl(rnk, j, 32);
        if (rj == lane) src = j;
    }
    float dsrt = __shfl(d,  src, 32);
    int   io   = __shfl(iv, src, 32);

    const float* gp = pts      + ((size_t)b * NSAMP + io) * 3;
    const float* gq = norm_ori + ((size_t)b * NSAMP + io) * 3;
    float gx = gp[0], gy = gp[1], gz = gp[2];
    float qx = gq[0], qy = gq[1], qz = gq[2];

    float lx = gx - cx, ly = gy - cy, lz = gz - cz;
    float len = sqrtf(lx*lx + ly*ly + lz*lz);
    float linv = (len == 0.f) ? 1.f : len;
    float ulx = lx/linv, uly = ly/linv, ulz = lz/linv;

    float a0 = ulx*nx + uly*ny + ulz*nz;
    float a1 = ulx*qx + uly*qy + ulz*qz;
    float dq = fminf(fmaxf(qx*nx + qy*ny + qz*nz, -1.f + EPSF), 1.f - EPSF);
    float an = acosf(dq);

    const int pl = (lane + 31) & 31;
    float lpx = __shfl(lx, pl, 32), lpy = __shfl(ly, pl, 32), lpz = __shfl(lz, pl, 32);
    float qpx = __shfl(qx, pl, 32), qpy = __shfl(qy, pl, 32), qpz = __shfl(qz, pl, 32);
    float dprev = __shfl(dsrt, pl, 32);

    float ix = lx - lpx, iy = ly - lpy, iz = lz - lpz;
    float il = sqrtf(ix*ix + iy*iy + iz*iz);
    float iinv = (il == 0.f) ? 1.f : il;
    float iux = ix/iinv, iuy = iy/iinv, iuz = iz/iinv;
    float i0 = iux*qx  + iuy*qy  + iuz*qz;
    float i1 = iux*qpx + iuy*qpy + iuz*qpz;
    float dqq = fminf(fmaxf(qx*qpx + qy*qpy + qz*qpz, -1.f + EPSF), 1.f - EPSF);
    float i2 = acosf(dqq);

    float pia = (lane == 0) ? (-3.f - dprev) : (dsrt - dprev);

    size_t p = ((size_t)b * 32 + lane) * 512 + s;   // pixel (b, h=sorted pos, w=s)
    __bf16* dst = X0 + p * 32;
    v8bf f0, z = {};
    f0[0] = (__bf16)len; f0[1] = (__bf16)pia; f0[2] = (__bf16)a0; f0[3] = (__bf16)a1;
    f0[4] = (__bf16)an;  f0[5] = (__bf16)i0;  f0[6] = (__bf16)i1; f0[7] = (__bf16)i2;
    *(v8bf*)(dst)      = f0;
    *(v8bf*)(dst + 8)  = z;
    *(v8bf*)(dst + 16) = z;
    *(v8bf*)(dst + 24) = z;
}

// ---------------------------------------------------------------------------
// GEMM (bf16 WMMA) + bias + BN-stat accumulation. CI is compile-time so the
// K-loop fully unrolls into back-to-back v_wmma with clause-batched b128 loads.
// Layouts: X [P][CI], W [co][CI], Y [P][co]. Wave tile: 16 ch x 16 px.
// Stats: per-lane ds_add_f32 into 16 per-channel LDS accumulators, one
// global flush per block (no shfl trees, no per-element waits).
// ---------------------------------------------------------------------------
template<int CI>
__global__ __launch_bounds__(256)
void gemm_bn_kernel(const __bf16* __restrict__ Xin,
                    const __bf16* __restrict__ Wb,
                    const float*  __restrict__ bias,
                    __bf16* __restrict__ Yout,
                    float* __restrict__ statSum, float* __restrict__ statSq,
                    int co)
{
    __shared__ float lsum[16];
    __shared__ float lsq[16];
    const int tid = threadIdx.x;
    if (tid < 16) { lsum[tid] = 0.f; lsq[tid] = 0.f; }
    __syncthreads();

    const int lane = tid & 31;
    const int wave = tid >> 5;
    const int h    = lane >> 4;
    const int l16  = lane & 15;
    const size_t pixBase = (size_t)blockIdx.x * 128 + (size_t)wave * 16;
    const int chBase = blockIdx.y * 16;

    const __bf16* wrow = Wb  + (size_t)(chBase + l16) * CI;   // A: M = l16
    const __bf16* xrow = Xin + (pixBase + l16) * (size_t)CI;  // B: N = l16

    v8f acc = {};
    #pragma unroll
    for (int k = 0; k < CI; k += 32) {
        // A fragment 16x32: lane half h holds K in [8h,8h+7] and [16+8h,23+8h]
        v8bf alo = *(const v8bf*)(wrow + k + 8*h);
        v8bf ahi = *(const v8bf*)(wrow + k + 16 + 8*h);
        // B fragment 32x16: lane half h holds K in [8h,8h+7] and [16+8h,23+8h]
        v8bf blo = *(const v8bf*)(xrow + k + 8*h);
        v8bf bhi = *(const v8bf*)(xrow + k + 16 + 8*h);
        v16bf a = __builtin_shufflevector(alo, ahi, 0,1,2,3,4,5,6,7,8,9,10,11,12,13,14,15);
        v16bf bb = __builtin_shufflevector(blo, bhi, 0,1,2,3,4,5,6,7,8,9,10,11,12,13,14,15);
        acc = __builtin_amdgcn_wmma_f32_16x16x32_bf16(
                  false, a, false, bb, (short)0, acc, false, false);
    }

    // D layout: VGPR r -> channel chBase + 8h + r, pixel pixBase + l16.
    // Bias for this lane is 8 contiguous floats -> two b128 loads.
    float4 b0 = *(const float4*)(bias + chBase + 8*h);
    float4 b1 = *(const float4*)(bias + chBase + 8*h + 4);
    float bl[8] = { b0.x, b0.y, b0.z, b0.w, b1.x, b1.y, b1.z, b1.w };

    v8bf yv;
    #pragma unroll
    for (int r = 0; r < 8; ++r) {
        int m  = 8*h + r;
        float y = acc[r] + bl[r];
        yv[r] = (__bf16)y;
        atomicAdd(&lsum[m], y);        // ds_add_f32, 16 distinct addresses
        atomicAdd(&lsq[m],  y * y);
    }
    // 8 contiguous bf16 channels -> single b128 store.
    *(v8bf*)(Yout + (pixBase + l16) * (size_t)co + chBase + 8*h) = yv;

    __syncthreads();
    if (tid < 16) {
        atomicAdd(&statSum[chBase + tid], lsum[tid]);
        atomicAdd(&statSq[chBase + tid],  lsq[tid]);
    }
}

// ---------------------------------------------------------------------------
// BN finalize: (sum, sumsq) -> per-channel scale/shift.
// ---------------------------------------------------------------------------
__global__ void bnfin_kernel(const float* __restrict__ sum, const float* __restrict__ sq,
                             const float* __restrict__ g, const float* __restrict__ beta,
                             float* __restrict__ scale, float* __restrict__ shift, int co)
{
    int c = blockIdx.x * blockDim.x + threadIdx.x;
    if (c >= co) return;
    const float invP = 1.0f / (float)PIXELS;
    float mean = sum[c] * invP;
    float var  = sq[c]  * invP - mean * mean;
    float iv   = rsqrtf(var + EPSBN);
    float sc   = g[c] * iv;
    scale[c] = sc;
    shift[c] = beta[c] - mean * sc;
}

// ---------------------------------------------------------------------------
// In-place BN apply + ReLU over Y [P][co]; 8 elements (16B) per thread.
// ---------------------------------------------------------------------------
__global__ __launch_bounds__(256)
void act_kernel(__bf16* __restrict__ Y, const float* __restrict__ scale,
                const float* __restrict__ shift, int co, size_t total)
{
    size_t i = ((size_t)blockIdx.x * 256 + threadIdx.x) * 8;
    if (i >= total) return;
    v8bf v = *(v8bf*)(Y + i);
    int c0 = (int)(i % (size_t)co);     // co multiple of 8: all 8 stay in-row
    #pragma unroll
    for (int e = 0; e < 8; ++e) {
        float f = fmaxf(scale[c0+e] * (float)v[e] + shift[c0+e], 0.f);
        v[e] = (__bf16)f;
    }
    *(v8bf*)(Y + i) = v;
}

// ---------------------------------------------------------------------------
// Final: apply last BN + ReLU, max over ns, write (B,S,384) f32. c-fastest
// thread mapping -> coalesced reads and stores.
// ---------------------------------------------------------------------------
__global__ __launch_bounds__(256)
void maxout_kernel(const __bf16* __restrict__ Y, const float* __restrict__ scale,
                   const float* __restrict__ shift, float* __restrict__ out)
{
    size_t i = (size_t)blockIdx.x * 256 + threadIdx.x;
    if (i >= (size_t)16 * 512 * 384) return;
    int c = (int)(i % 384);
    int s = (int)((i / 384) % 512);
    int b = (int)(i / (384 * 512));
    float sc = scale[c], sh = shift[c];
    float mx = -INFINITY;
    for (int n = 0; n < 32; ++n) {
        float y = (float)Y[(((size_t)b * 32 + n) * 512 + s) * 384 + c];
        mx = fmaxf(mx, fmaxf(sc * y + sh, 0.f));
    }
    out[i] = mx;    // out[(b*512+s)*384 + c]
}

// ---------------------------------------------------------------------------
// Utilities: zero stats, convert/pad f32 weights to bf16.
// ---------------------------------------------------------------------------
__global__ void zero_kernel(float* p, int n)
{
    int i = blockIdx.x * 256 + threadIdx.x;
    if (i < n) p[i] = 0.f;
}

__global__ void wconv_kernel(const float* __restrict__ W, __bf16* __restrict__ Wb,
                             int co, int ci_in, int ci_out)
{
    int i = blockIdx.x * 256 + threadIdx.x;
    if (i >= co * ci_out) return;
    int r = i / ci_out, k = i % ci_out;
    Wb[i] = (k < ci_in) ? (__bf16)W[r * ci_in + k] : (__bf16)0.f;
}

// ---------------------------------------------------------------------------
extern "C" void kernel_launch(void* const* d_in, const int* in_sizes, int n_in,
                              void* d_out, int out_size, void* d_ws, size_t ws_size,
                              hipStream_t stream)
{
    (void)in_sizes; (void)n_in; (void)out_size; (void)ws_size;
    const float* xyz      = (const float*)d_in[0];
    const float* centers  = (const float*)d_in[1];
    const float* cnorms   = (const float*)d_in[2];
    const int*   idx      = (const int*)d_in[3];
    const float* pts      = (const float*)d_in[4];
    const float* norm_ori = (const float*)d_in[5];
    const float* convW[4] = { (const float*)d_in[6],  (const float*)d_in[7],
                              (const float*)d_in[8],  (const float*)d_in[9]  };
    const float* convB[4] = { (const float*)d_in[10], (const float*)d_in[11],
                              (const float*)d_in[12], (const float*)d_in[13] };
    const float* bnG[4]   = { (const float*)d_in[14], (const float*)d_in[15],
                              (const float*)d_in[16], (const float*)d_in[17] };
    const float* bnB[4]   = { (const float*)d_in[18], (const float*)d_in[19],
                              (const float*)d_in[20], (const float*)d_in[21] };
    float* out = (float*)d_out;

    // ---- workspace layout (in-place act keeps peak ~321 MB) ----
    char* ws = (char*)d_ws;
    size_t off = 0;
    auto alloc = [&](size_t bytes) -> void* {
        void* p = ws + off; off += (bytes + 255) & ~(size_t)255; return p;
    };
    __bf16* arenaA = (__bf16*)alloc(192ull << 20);  // X0(16M) -> Y2(64M) -> Y4(192M)
    __bf16* arenaB = (__bf16*)alloc(128ull << 20);  // Y1(32M) -> Y3(128M)

    const int ciPad[4] = { 32, 64, 128, 256 };
    const int ciIn[4]  = {  8, 64, 128, 256 };
    const int coL[4]   = { 64, 128, 256, 384 };
    __bf16* Wb[4];
    for (int l = 0; l < 4; ++l) Wb[l] = (__bf16*)alloc((size_t)coL[l] * ciPad[l] * 2);

    float* stats = (float*)alloc(3328 * 4);   // [sum|sq] per layer then [scale|shift]
    float *sum[4], *sq[4], *scale[4], *shift[4];
    {
        float* p = stats;
        for (int l = 0; l < 4; ++l) { sum[l] = p; p += coL[l]; sq[l] = p; p += coL[l]; }
        for (int l = 0; l < 4; ++l) { scale[l] = p; p += coL[l]; shift[l] = p; p += coL[l]; }
    }

    __bf16* X0 = arenaA;
    __bf16* Y1 = arenaB;
    __bf16* Y2 = arenaA;
    __bf16* Y3 = arenaB;
    __bf16* Y4 = arenaA;

    // ---- pipeline ----
    zero_kernel<<<(1664 + 255) / 256, 256, 0, stream>>>(stats, 1664);
    for (int l = 0; l < 4; ++l) {
        int n = coL[l] * ciPad[l];
        wconv_kernel<<<(n + 255) / 256, 256, 0, stream>>>(convW[l], Wb[l],
                                                          coL[l], ciIn[l], ciPad[l]);
    }

    ri_kernel<<<2048, 128, 0, stream>>>(xyz, centers, cnorms, idx, pts, norm_ori, X0);

    const int pixBlocks = (int)(PIXELS / 128);   // 2048
    // layer 1: 32(padded) -> 64
    gemm_bn_kernel<32><<<dim3(pixBlocks, coL[0] / 16), 256, 0, stream>>>(
        X0, Wb[0], convB[0], Y1, sum[0], sq[0], coL[0]);
    bnfin_kernel<<<1, coL[0], 0, stream>>>(sum[0], sq[0], bnG[0], bnB[0],
                                           scale[0], shift[0], coL[0]);
    act_kernel<<<(int)(PIXELS * coL[0] / 8 / 256), 256, 0, stream>>>(
        Y1, scale[0], shift[0], coL[0], PIXELS * coL[0]);
    // layer 2: 64 -> 128
    gemm_bn_kernel<64><<<dim3(pixBlocks, coL[1] / 16), 256, 0, stream>>>(
        Y1, Wb[1], convB[1], Y2, sum[1], sq[1], coL[1]);
    bnfin_kernel<<<1, coL[1], 0, stream>>>(sum[1], sq[1], bnG[1], bnB[1],
                                           scale[1], shift[1], coL[1]);
    act_kernel<<<(int)(PIXELS * coL[1] / 8 / 256), 256, 0, stream>>>(
        Y2, scale[1], shift[1], coL[1], PIXELS * coL[1]);
    // layer 3: 128 -> 256
    gemm_bn_kernel<128><<<dim3(pixBlocks, coL[2] / 16), 256, 0, stream>>>(
        Y2, Wb[2], convB[2], Y3, sum[2], sq[2], coL[2]);
    bnfin_kernel<<<1, coL[2], 0, stream>>>(sum[2], sq[2], bnG[2], bnB[2],
                                           scale[2], shift[2], coL[2]);
    act_kernel<<<(int)(PIXELS * coL[2] / 8 / 256), 256, 0, stream>>>(
        Y3, scale[2], shift[2], coL[2], PIXELS * coL[2]);
    // layer 4: 256 -> 384
    gemm_bn_kernel<256><<<dim3(pixBlocks, coL[3] / 16), 256, 0, stream>>>(
        Y3, Wb[3], convB[3], Y4, sum[3], sq[3], coL[3]);
    bnfin_kernel<<<1, coL[3], 0, stream>>>(sum[3], sq[3], bnG[3], bnB[3],
                                           scale[3], shift[3], coL[3]);

    // BN4 + ReLU + max over ns -> (B, S, 384)
    maxout_kernel<<<(int)(((size_t)16 * 512 * 384 + 255) / 256), 256, 0, stream>>>(
        Y4, scale[3], shift[3], out);
}
